// ResidualSelfAttentionCell_28269474742740
// MI455X (gfx1250) — matrix operand order repaired and checked by reading, not compiled, not run on hardware
//
#include <hip/hip_runtime.h>
#include <math.h>

// ---------------------------------------------------------------------------
// ResidualSelfAttentionCell for MI455X (gfx1250, wave32, WMMA bf16 16x16x32)
// B=8, N=1024, F=512, E=512, H=8, DH=64, LOGIT_SCALE=8
// ---------------------------------------------------------------------------

#define Bc  8
#define Nc  1024
#define Fc  512
#define Ec  512
#define Hc  8
#define DHc 64
#define Mc  (Bc * Nc)          // 8192 rows
#define INV_LOGIT 0.125f       // 1/sqrt(E/H) = 1/8

typedef float  f32x8   __attribute__((ext_vector_type(8)));
typedef __bf16 bf16x16 __attribute__((ext_vector_type(16)));
typedef __bf16 bf16x8  __attribute__((ext_vector_type(8)));

union Frag { bf16x16 v; bf16x8 h[2]; };

__device__ __forceinline__ f32x8 wmma_bf16(bf16x16 a, bf16x16 b, f32x8 c) {
  // D = A(16x32 bf16) * B(32x16 bf16) + C(16x16 f32)
  return __builtin_amdgcn_wmma_f32_16x16x32_bf16(
      /*neg_a=*/false, a, /*neg_b=*/false, b,
      /*c_mod=*/(short)0, c, /*reuse_a=*/false, /*reuse_b=*/false);
}

__device__ __forceinline__ float red16_max(float v) {
  // xor masks < 16 keep the reduction inside each 16-lane half (rows live there)
  v = fmaxf(v, __shfl_xor(v, 1, 32));
  v = fmaxf(v, __shfl_xor(v, 2, 32));
  v = fmaxf(v, __shfl_xor(v, 4, 32));
  v = fmaxf(v, __shfl_xor(v, 8, 32));
  return v;
}
__device__ __forceinline__ float red16_sum(float v) {
  v += __shfl_xor(v, 1, 32);
  v += __shfl_xor(v, 2, 32);
  v += __shfl_xor(v, 4, 32);
  v += __shfl_xor(v, 8, 32);
  return v;
}

// ---------------------------------------------------------------------------
// f32 -> bf16 cast (weights)
// ---------------------------------------------------------------------------
__global__ void cast_f32_to_bf16(const float* __restrict__ s,
                                 __bf16* __restrict__ d, int n) {
  int i = blockIdx.x * blockDim.x + threadIdx.x;
  if (i < n) d[i] = (__bf16)s[i];
}

// ---------------------------------------------------------------------------
// LayerNorm(inp) -> bf16  AND  raw inp -> bf16 (V projection uses raw input)
// One block per row of 512, 256 threads x float2.
// ---------------------------------------------------------------------------
__global__ __launch_bounds__(256)
void ln1_cast_kernel(const float* __restrict__ inp,
                     const float* __restrict__ g, const float* __restrict__ bb,
                     __bf16* __restrict__ xln, __bf16* __restrict__ inpb) {
  int row = blockIdx.x;
  const float2 v = ((const float2*)(inp + (size_t)row * Fc))[threadIdx.x];
  float s = v.x + v.y, q = v.x * v.x + v.y * v.y;
  for (int m = 16; m; m >>= 1) { s += __shfl_xor(s, m, 32); q += __shfl_xor(q, m, 32); }
  __shared__ float sb[8], qb[8];
  int w = threadIdx.x >> 5;
  if ((threadIdx.x & 31) == 0) { sb[w] = s; qb[w] = q; }
  __syncthreads();
  float S = 0.f, Q = 0.f;
  for (int i = 0; i < 8; i++) { S += sb[i]; Q += qb[i]; }
  float mu  = S * (1.f / Fc);
  float var = Q * (1.f / Fc) - mu * mu;
  float r   = rsqrtf(var + 1e-5f);
  int c = threadIdx.x * 2;
  size_t o = (size_t)row * Fc + c;
  xln[o]     = (__bf16)((v.x - mu) * r * g[c]     + bb[c]);
  xln[o + 1] = (__bf16)((v.y - mu) * r * g[c + 1] + bb[c + 1]);
  inpb[o]     = (__bf16)v.x;
  inpb[o + 1] = (__bf16)v.y;
}

// ---------------------------------------------------------------------------
// Final LayerNorm: out_pre (f32) -> d_out (f32)
// ---------------------------------------------------------------------------
__global__ __launch_bounds__(256)
void ln2_kernel(const float* __restrict__ xin,
                const float* __restrict__ g, const float* __restrict__ bb,
                float* __restrict__ out) {
  int row = blockIdx.x;
  const float2 v = ((const float2*)(xin + (size_t)row * Fc))[threadIdx.x];
  float s = v.x + v.y, q = v.x * v.x + v.y * v.y;
  for (int m = 16; m; m >>= 1) { s += __shfl_xor(s, m, 32); q += __shfl_xor(q, m, 32); }
  __shared__ float sb[8], qb[8];
  int w = threadIdx.x >> 5;
  if ((threadIdx.x & 31) == 0) { sb[w] = s; qb[w] = q; }
  __syncthreads();
  float S = 0.f, Q = 0.f;
  for (int i = 0; i < 8; i++) { S += sb[i]; Q += qb[i]; }
  float mu  = S * (1.f / Fc);
  float var = Q * (1.f / Fc) - mu * mu;
  float r   = rsqrtf(var + 1e-5f);
  int c = threadIdx.x * 2;
  size_t o = (size_t)row * Fc + c;
  out[o]     = (v.x - mu) * r * g[c]     + bb[c];
  out[o + 1] = (v.y - mu) * r * g[c + 1] + bb[c + 1];
}

// ---------------------------------------------------------------------------
// WMMA GEMM: C[M,O] = A_bf16[M,K] @ W_bf16[O,K]^T  (f32 accum)
// Block = 128 threads (4 waves); each wave computes a 16(M) x 64(O) strip.
// Grid: (M/64, O/64).  K-loop is software-pipelined: next iteration's A+B
// fragments are in flight while the current 4 WMMAs issue, so the scheduler
// waits on the older load group instead of draining loadcnt to 0.
// Epilogues:
//   mode 0: QK projection -> de-interleave to q[B,H,N,DH] (scaled) / k[B,H,N,DH]
//   mode 1: V  projection -> transposed scatter to vT[B,H,DH,N]
//   mode 2: post proj     -> out_pre = inp + acc + bias   (f32)
// ---------------------------------------------------------------------------
__global__ __launch_bounds__(128)
void gemm_bf16_kernel(const __bf16* __restrict__ A, const __bf16* __restrict__ W,
                      const float* __restrict__ bias, int K, int mode,
                      __bf16* __restrict__ q_out, __bf16* __restrict__ k_out,
                      __bf16* __restrict__ vt_out,
                      float* __restrict__ out_pre, const float* __restrict__ inp_raw) {
  const int lane = threadIdx.x & 31;
  const int hf   = lane >> 4;        // lane half (0: lanes 0-15, 1: lanes 16-31)
  const int l16  = lane & 15;
  const int m0   = blockIdx.x * 64 + (threadIdx.x >> 5) * 16;
  const int o0   = blockIdx.y * 64;

  f32x8 acc[4] = {};
  // A fragment (16x32 bf16): lane half 0 reads K {k0..k0+7, k0+16..k0+23},
  // half 1 reads K {k0+8..k0+15, k0+24..k0+31}; row = m0 + l16.
  const __bf16* arow  = A + (size_t)(m0 + l16) * K + hf * 8;
  // B fragment (32x16 bf16): lane n = l16, same K order as the A fragment.
  const __bf16* wbase = W + (size_t)(o0 + l16) * K + hf * 8;

  // ---- prologue: load fragments for k0 = 0 --------------------------------
  Frag a_cur, b_cur[4];
  a_cur.h[0] = *(const bf16x8*)(arow);
  a_cur.h[1] = *(const bf16x8*)(arow + 16);
#pragma unroll
  for (int t = 0; t < 4; t++) {
    const __bf16* wr = wbase + (size_t)t * 16 * K;
    b_cur[t].h[0] = *(const bf16x8*)(wr);
    b_cur[t].h[1] = *(const bf16x8*)(wr + 16);
  }

  // ---- pipelined main loop -------------------------------------------------
  for (int k0 = 0; k0 + 32 < K; k0 += 32) {
    const int kn = k0 + 32;
    __builtin_prefetch(arow + kn + 96, 0, 3);   // global_prefetch_b8, near-cache
    Frag a_nxt, b_nxt[4];
    a_nxt.h[0] = *(const bf16x8*)(arow + kn);
    a_nxt.h[1] = *(const bf16x8*)(arow + kn + 16);
#pragma unroll
    for (int t = 0; t < 4; t++) {
      const __bf16* wr = wbase + (size_t)t * 16 * K + kn;
      b_nxt[t].h[0] = *(const bf16x8*)(wr);
      b_nxt[t].h[1] = *(const bf16x8*)(wr + 16);
    }
#pragma unroll
    for (int t = 0; t < 4; t++)
      acc[t] = wmma_bf16(a_cur.v, b_cur[t].v, acc[t]);
    a_cur = a_nxt;
#pragma unroll
    for (int t = 0; t < 4; t++) b_cur[t] = b_nxt[t];
  }
  // ---- epilogue WMMAs ------------------------------------------------------
#pragma unroll
  for (int t = 0; t < 4; t++)
    acc[t] = wmma_bf16(a_cur.v, b_cur[t].v, acc[t]);

  // C/D layout: VGPR j, lane L -> row (j + 8*(L>=16)), col L%16.
#pragma unroll
  for (int t = 0; t < 4; t++) {
#pragma unroll
    for (int j = 0; j < 8; j++) {
      int m = m0 + j + hf * 8;
      int o = o0 + t * 16 + l16;
      float val = acc[t][j] + bias[o];
      if (mode == 0) {
        int b = m >> 10, n = m & (Nc - 1);
        int h = o >> 7, r = o & 127, dh = r >> 1;
        size_t idx = (((size_t)b * Hc + h) * Nc + n) * DHc + dh;
        if ((r & 1) == 0) q_out[idx] = (__bf16)(val * INV_LOGIT);
        else              k_out[idx] = (__bf16)val;
      } else if (mode == 1) {
        int b = m >> 10, n = m & (Nc - 1);
        int h = o >> 6, dh = o & 63;
        vt_out[(((size_t)b * Hc + h) * DHc + dh) * Nc + n] = (__bf16)val;
      } else {
        size_t idx = (size_t)m * Fc + o;
        out_pre[idx] = inp_raw[idx] + val;
      }
    }
  }
}

// ---------------------------------------------------------------------------
// Flash attention per (b,h): grid (B*H, N/64), block 128 (4 waves).
// Each wave owns 16 query rows; keys streamed 32 at a time.
// q is pre-scaled by 1/LOGIT_SCALE; v is stored transposed vT[B,H,DH,N].
// P is round-tripped through LDS to convert the C/D layout to the A layout.
// ---------------------------------------------------------------------------
__global__ __launch_bounds__(128)
void attn_kernel(const __bf16* __restrict__ q, const __bf16* __restrict__ kmat,
                 const __bf16* __restrict__ vt, __bf16* __restrict__ att) {
  const int bh   = blockIdx.x;                 // b*H + h
  const int b    = bh >> 3, h = bh & 7;
  const int wave = threadIdx.x >> 5;
  const int n0   = blockIdx.y * 64 + wave * 16;
  const int lane = threadIdx.x & 31;
  const int hf   = lane >> 4, l16 = lane & 15;

  __shared__ __align__(16) __bf16 pshm[4][16][40];   // padded rows (80B) vs bank conflicts
  __bf16 (*pw)[40] = pshm[wave];

  // Q A-fragments for dh 0..31 and 32..63 (stay in registers).
  const __bf16* qrow = q + (((size_t)bh) * Nc + n0 + l16) * DHc + hf * 8;
  Frag qf[2];
  qf[0].h[0] = *(const bf16x8*)(qrow + 0);
  qf[0].h[1] = *(const bf16x8*)(qrow + 16);
  qf[1].h[0] = *(const bf16x8*)(qrow + 32);
  qf[1].h[1] = *(const bf16x8*)(qrow + 48);

  f32x8 oacc[4] = {};
  float mrow[8], lrow[8];
#pragma unroll
  for (int j = 0; j < 8; j++) { mrow[j] = -3.0e38f; lrow[j] = 0.f; }

  const __bf16* kb = kmat + (size_t)bh * Nc * DHc;
  const __bf16* vb = vt   + (size_t)bh * DHc * Nc;

  for (int j0 = 0; j0 < Nc; j0 += 32) {
    // ---- S = Q @ K^T for 32 keys: two 16x16 tiles --------------------------
    f32x8 c[2] = {};
#pragma unroll
    for (int t = 0; t < 2; t++) {
      const __bf16* krow = kb + (size_t)(j0 + t * 16 + l16) * DHc + hf * 8;
#pragma unroll
      for (int s = 0; s < 2; s++) {
        Frag bf;
        bf.h[0] = *(const bf16x8*)(krow + s * 32);
        bf.h[1] = *(const bf16x8*)(krow + s * 32 + 16);
        c[t] = wmma_bf16(qf[s].v, bf.v, c[t]);
      }
    }

    // ---- online softmax update --------------------------------------------
    float alpha[8];
#pragma unroll
    for (int j = 0; j < 8; j++) {
      float tmax = red16_max(fmaxf(c[0][j], c[1][j]));
      float mn   = fmaxf(mrow[j], tmax);
      alpha[j]   = __expf(mrow[j] - mn);
      mrow[j]    = mn;
    }
#pragma unroll
    for (int j = 0; j < 8; j++) {
      c[0][j] = __expf(c[0][j] - mrow[j]);
      c[1][j] = __expf(c[1][j] - mrow[j]);
      float s = red16_sum(c[0][j] + c[1][j]);
      lrow[j] = lrow[j] * alpha[j] + s;
    }
#pragma unroll
    for (int t = 0; t < 4; t++)
#pragma unroll
      for (int j = 0; j < 8; j++) oacc[t][j] *= alpha[j];

    // ---- P: C/D layout -> LDS -> A layout ---------------------------------
#pragma unroll
    for (int j = 0; j < 8; j++) {
      int r = j + hf * 8;
      pw[r][l16]      = (__bf16)c[0][j];   // keys j0+0..15
      pw[r][16 + l16] = (__bf16)c[1][j];   // keys j0+16..31
    }
    __syncthreads();   // DS in-order within wave; barrier for extra safety
    Frag pf;
    const __bf16* pr = &pw[l16][hf * 8];
    pf.h[0] = *(const bf16x8*)(pr);
    pf.h[1] = *(const bf16x8*)(pr + 16);

    // ---- O += P @ V (V transposed: K-contiguous B fragments) --------------
#pragma unroll
    for (int t = 0; t < 4; t++) {
      const __bf16* vrow = vb + (size_t)(t * 16 + l16) * Nc + j0 + hf * 8;
      Frag bf;
      bf.h[0] = *(const bf16x8*)(vrow);
      bf.h[1] = *(const bf16x8*)(vrow + 16);
      oacc[t] = wmma_bf16(pf.v, bf.v, oacc[t]);
    }
    __syncthreads();
  }

  // ---- normalize and write att[b,n, h*64+dh] as bf16 ----------------------
#pragma unroll
  for (int t = 0; t < 4; t++) {
#pragma unroll
    for (int j = 0; j < 8; j++) {
      int n = n0 + j + hf * 8;
      int e = h * DHc + t * 16 + l16;
      att[((size_t)b * Nc + n) * Ec + e] = (__bf16)(oacc[t][j] / lrow[j]);
    }
  }
}

// ---------------------------------------------------------------------------
// Host-side orchestration
// ---------------------------------------------------------------------------
extern "C" void kernel_launch(void* const* d_in, const int* in_sizes, int n_in,
                              void* d_out, int out_size, void* d_ws, size_t ws_size,
                              hipStream_t stream) {
  (void)in_sizes; (void)n_in; (void)out_size; (void)ws_size;
  const float* inp    = (const float*)d_in[0];
  const float* qk_w   = (const float*)d_in[1];
  const float* qk_b   = (const float*)d_in[2];
  const float* v_w    = (const float*)d_in[3];
  const float* v_b    = (const float*)d_in[4];
  const float* post_w = (const float*)d_in[5];
  const float* post_b = (const float*)d_in[6];
  const float* ln1_g  = (const float*)d_in[7];
  const float* ln1_b  = (const float*)d_in[8];
  const float* ln2_g  = (const float*)d_in[9];
  const float* ln2_b  = (const float*)d_in[10];
  float* out = (float*)d_out;

  char* ws = (char*)d_ws;
  size_t off = 0;
  auto carve = [&](size_t bytes) -> char* {
    char* p = ws + off;
    off += (bytes + 255) & ~(size_t)255;
    return p;
  };
  __bf16* xln     = (__bf16*)carve((size_t)Mc * Fc * 2);        // LN1(inp) bf16
  __bf16* inpb    = (__bf16*)carve((size_t)Mc * Fc * 2);        // raw inp bf16
  __bf16* qkw_b   = (__bf16*)carve((size_t)2 * Ec * Fc * 2);
  __bf16* vw_b    = (__bf16*)carve((size_t)Ec * Fc * 2);
  __bf16* pw_b    = (__bf16*)carve((size_t)Fc * Ec * 2);
  __bf16* qb      = (__bf16*)carve((size_t)Bc * Hc * Nc * DHc * 2);
  __bf16* kbuf    = (__bf16*)carve((size_t)Bc * Hc * Nc * DHc * 2);
  __bf16* vtb     = (__bf16*)carve((size_t)Bc * Hc * DHc * Nc * 2);
  __bf16* attb    = (__bf16*)carve((size_t)Mc * Ec * 2);
  float*  out_pre = (float*)carve((size_t)Mc * Fc * 4);

  // 1) cast weights to bf16
  cast_f32_to_bf16<<<(2 * Ec * Fc + 255) / 256, 256, 0, stream>>>(qk_w, qkw_b, 2 * Ec * Fc);
  cast_f32_to_bf16<<<(Ec * Fc + 255) / 256, 256, 0, stream>>>(v_w, vw_b, Ec * Fc);
  cast_f32_to_bf16<<<(Fc * Ec + 255) / 256, 256, 0, stream>>>(post_w, pw_b, Fc * Ec);

  // 2) LN1 + input cast
  ln1_cast_kernel<<<Mc, 256, 0, stream>>>(inp, ln1_g, ln1_b, xln, inpb);

  // 3) QK projection (M=8192, O=1024, K=512) -> q (scaled), k
  gemm_bf16_kernel<<<dim3(Mc / 64, (2 * Ec) / 64), 128, 0, stream>>>(
      xln, qkw_b, qk_b, Fc, 0, qb, kbuf, nullptr, nullptr, nullptr);

  // 4) V projection (uses RAW input) -> vT[B,H,DH,N]
  gemm_bf16_kernel<<<dim3(Mc / 64, Ec / 64), 128, 0, stream>>>(
      inpb, vw_b, v_b, Fc, 1, nullptr, nullptr, vtb, nullptr, nullptr);

  // 5) attention -> att[B,N,E] bf16
  attn_kernel<<<dim3(Bc * Hc, Nc / 64), 128, 0, stream>>>(qb, kbuf, vtb, attb);

  // 6) post projection + residual -> out_pre f32
  gemm_bf16_kernel<<<dim3(Mc / 64, Fc / 64), 128, 0, stream>>>(
      attb, pw_b, post_b, Ec, 2, nullptr, nullptr, nullptr, out_pre, inp);

  // 7) LN2 -> d_out
  ln2_kernel<<<Mc, 256, 0, stream>>>(out_pre, ln2_g, ln2_b, out);
}
